// EncoderLayer_69836168233598
// MI455X (gfx1250) — matrix operand order repaired and checked by reading, not compiled, not run on hardware
//
#include <hip/hip_runtime.h>
#include <hip/hip_bf16.h>
#include <stdint.h>

// ---------------------------------------------------------------------------
// Transformer encoder layer for MI455X (gfx1250, wave32, WMMA).
// bf16 WMMA (f32 accum) everywhere. Attention is fused flash-style BUT still
// materializes the softmax probabilities (required output) with exactly ONE
// non-temporal HBM write pass; Q/K/V re-reads stay inside the 192MB L2.
// ---------------------------------------------------------------------------

typedef __attribute__((ext_vector_type(16))) __bf16 v16bf;
typedef __attribute__((ext_vector_type(8)))  __bf16 v8bf;
typedef __attribute__((ext_vector_type(8)))  float  v8f;

// Problem dims (fixed by the reference)
constexpr int Bd  = 2;
constexpr int Sd  = 2048;
constexpr int Dd  = 1024;
constexpr int Fd  = 4096;
constexpr int Hd  = 16;
constexpr int HDd = 64;          // head dim
constexpr int BSd = Bd * Sd;     // 4096 rows
constexpr int BHd = Bd * Hd;     // 32 (b,h) batches

// GEMM tiling
#define BM 64
#define BN 64
#define BK 32
#define LDSS 40   // padded LDS row stride (elements)

static __device__ __forceinline__ v16bf pack16(v8bf lo, v8bf hi) {
    return __builtin_shufflevector(lo, hi, 0,1,2,3,4,5,6,7,8,9,10,11,12,13,14,15);
}

// ---------------------------------------------------------------------------
// Tiled GEMM: C[M,N] (+bias, optional relu) = A[M,K] * Bt[N,K]^T   (all bf16 in)
// blockDim = 128 (4 waves); wave w owns rows [w*16, w*16+16) of the 64x64 tile.
// ---------------------------------------------------------------------------
__global__ __launch_bounds__(128) void gemm_bt(
    const __bf16* __restrict__ A, const __bf16* __restrict__ Bt,
    const float* __restrict__ bias,
    float* __restrict__ C, __bf16* __restrict__ Cbf,
    int M, int N, int K, int relu)
{
    __shared__ __align__(16) __bf16 As[BM * LDSS];
    __shared__ __align__(16) __bf16 Bs[BN * LDSS];

    const int t    = threadIdx.x;
    const int lane = t & 31;
    const int wave = t >> 5;
    const int m0 = blockIdx.y * BM;
    const int n0 = blockIdx.x * BN;

    v8f acc[4] = {};
    const int lr    = lane & 15;   // row within 16-row fragment
    const int khalf = lane >> 4;   // lane half selects k-chunks

    for (int kk = 0; kk < K; kk += BK) {
        #pragma unroll
        for (int i = 0; i < 2; ++i) {
            int chunk = t + i * 128;
            int row   = chunk >> 2;
            int c8    = (chunk & 3) * 8;
            *(v8bf*)&As[row * LDSS + c8] =
                *(const v8bf*)(A + (long long)(m0 + row) * K + kk + c8);
            *(v8bf*)&Bs[row * LDSS + c8] =
                *(const v8bf*)(Bt + (long long)(n0 + row) * K + kk + c8);
        }
        __syncthreads();

        int ar = wave * 16 + lr;
        v16bf afrag = pack16(*(const v8bf*)&As[ar * LDSS + khalf * 8],
                             *(const v8bf*)&As[ar * LDSS + khalf * 8 + 16]);
        #pragma unroll
        for (int j = 0; j < 4; ++j) {
            int br = j * 16 + lr;
            v16bf bfrag = pack16(*(const v8bf*)&Bs[br * LDSS + khalf * 8],
                                 *(const v8bf*)&Bs[br * LDSS + khalf * 8 + 16]);
            acc[j] = __builtin_amdgcn_wmma_f32_16x16x32_bf16(
                false, afrag, false, bfrag, (short)0, acc[j], false, false);
        }
        __syncthreads();
    }

    const int orow0 = m0 + wave * 16 + khalf * 8;
    #pragma unroll
    for (int j = 0; j < 4; ++j) {
        int col = n0 + j * 16 + lr;
        float bv = bias ? bias[col] : 0.0f;
        #pragma unroll
        for (int r = 0; r < 8; ++r) {
            float v = acc[j][r] + bv;
            if (relu) v = fmaxf(v, 0.0f);
            long long off = (long long)(orow0 + r) * N + col;
            if (C)   C[off]   = v;
            if (Cbf) Cbf[off] = (__bf16)v;
        }
    }
}

// ---------------------------------------------------------------------------
// Fused attention: per block = one (b,h) and one 16-row query block.
//   phase 1: scores = Q K^T / 8 (WMMA, K fragments straight from L2-resident
//            global), mask, row softmax across 2048 cols held in registers,
//            single non-temporal write of probs to d_out.
//   phase 2: ctx = probs @ V. Prob tiles (D-layout) round-trip through
//            wave-private LDS to become A-fragments; V fragments from global
//            (L2-resident). Cross-wave reduction in LDS; ctx written directly
//            in merged (b, s, h*64+d) bf16 layout.
// blockDim = 256 (8 waves); wave w owns score columns [w*256, w*256+256).
// ---------------------------------------------------------------------------
__global__ __launch_bounds__(256) void attn_fused(
    const __bf16* __restrict__ Qh,   // (b,h,s,64)
    const __bf16* __restrict__ Kh,   // (b,h,t,64)
    const __bf16* __restrict__ Vt,   // (b,h,64,t)
    const unsigned char* __restrict__ mask,  // (b,1,s,t)
    float* __restrict__ attn,        // (b,h,s,t)
    __bf16* __restrict__ ctxbf)      // (b,s,D)
{
    __shared__ float red[16 * 128];        // 8KB  row partials
    __shared__ float red2[16 * 16];        // 1KB
    __shared__ float rowmax[16];
    __shared__ float rowinv[16];
    __shared__ __align__(16) __bf16 Pst[8 * 16 * 40];   // 10KB wave-private prob tiles
    __shared__ float ctxred[8 * 16 * 64];  // 32KB cross-wave ctx reduction

    const int t    = threadIdx.x;
    const int lane = t & 31;
    const int wave = t >> 5;
    const int lr   = lane & 15;
    const int kh   = lane >> 4;

    const int bh   = blockIdx.y;
    const int bidx = bh >> 4;      // b   (H = 16)
    const int hidx = bh & 15;      // h
    const int s0   = blockIdx.x * 16;

    // ---- phase 1: scores ---------------------------------------------------
    // Q A-fragments for this 16-row block (k = 0..31 and 32..63)
    const __bf16* qp = Qh + ((long long)(bh * Sd + s0 + lr)) * HDd;
    v16bf qf0 = pack16(*(const v8bf*)(qp + kh * 8),      *(const v8bf*)(qp + kh * 8 + 16));
    v16bf qf1 = pack16(*(const v8bf*)(qp + 32 + kh * 8), *(const v8bf*)(qp + 32 + kh * 8 + 16));

    v8f acc[16] = {};
    #pragma unroll
    for (int j = 0; j < 16; ++j) {
        const __bf16* kp = Kh + ((long long)(bh * Sd + wave * 256 + j * 16 + lr)) * HDd;
        v16bf bf0 = pack16(*(const v8bf*)(kp + kh * 8),      *(const v8bf*)(kp + kh * 8 + 16));
        v16bf bf1 = pack16(*(const v8bf*)(kp + 32 + kh * 8), *(const v8bf*)(kp + 32 + kh * 8 + 16));
        acc[j] = __builtin_amdgcn_wmma_f32_16x16x32_bf16(
            false, qf0, false, bf0, (short)0, acc[j], false, false);
        acc[j] = __builtin_amdgcn_wmma_f32_16x16x32_bf16(
            false, qf1, false, bf1, (short)0, acc[j], false, false);
    }

    // scale + mask + per-lane row max (lane holds rows r + 8*kh, r = 0..7)
    float pmax[8];
    #pragma unroll
    for (int r = 0; r < 8; ++r) pmax[r] = -3.0e38f;
    #pragma unroll
    for (int r = 0; r < 8; ++r) {
        const unsigned char* mrow =
            mask + ((long long)(bidx * Sd + s0 + r + 8 * kh)) * Sd;
        #pragma unroll
        for (int j = 0; j < 16; ++j) {
            int c = wave * 256 + j * 16 + lr;
            float v = acc[j][r] * 0.125f;              // / sqrt(64)
            if (mrow[c]) v = -1.0e10f;
            acc[j][r] = v;
            pmax[r] = fmaxf(pmax[r], v);
        }
    }

    // block-wide row max
    #pragma unroll
    for (int r = 0; r < 8; ++r) red[(r + 8 * kh) * 128 + wave * 16 + lr] = pmax[r];
    __syncthreads();
    {
        int row = t >> 4, i = t & 15;
        float m = red[row * 128 + i];
        #pragma unroll
        for (int q = 1; q < 8; ++q) m = fmaxf(m, red[row * 128 + i + 16 * q]);
        red2[row * 16 + i] = m;
    }
    __syncthreads();
    if (t < 16) {
        float m = red2[t * 16];
        for (int i = 1; i < 16; ++i) m = fmaxf(m, red2[t * 16 + i]);
        rowmax[t] = m;
    }
    __syncthreads();

    // exp + per-lane row sum
    float psum[8];
    #pragma unroll
    for (int r = 0; r < 8; ++r) psum[r] = 0.0f;
    #pragma unroll
    for (int r = 0; r < 8; ++r) {
        float mx = rowmax[r + 8 * kh];
        #pragma unroll
        for (int j = 0; j < 16; ++j) {
            float e = __expf(acc[j][r] - mx);
            acc[j][r] = e;
            psum[r] += e;
        }
    }
    #pragma unroll
    for (int r = 0; r < 8; ++r) red[(r + 8 * kh) * 128 + wave * 16 + lr] = psum[r];
    __syncthreads();
    {
        int row = t >> 4, i = t & 15;
        float s = red[row * 128 + i];
        #pragma unroll
        for (int q = 1; q < 8; ++q) s += red[row * 128 + i + 16 * q];
        red2[row * 16 + i] = s;
    }
    __syncthreads();
    if (t < 16) {
        float s = red2[t * 16];
        for (int i = 1; i < 16; ++i) s += red2[t * 16 + i];
        rowinv[t] = 1.0f / s;
    }
    __syncthreads();

    // normalize + single non-temporal write of the probability block
    #pragma unroll
    for (int r = 0; r < 8; ++r) {
        float inv = rowinv[r + 8 * kh];
        float* arow = attn + ((long long)bh * Sd + s0 + r + 8 * kh) * Sd;
        #pragma unroll
        for (int j = 0; j < 16; ++j) {
            float p = acc[j][r] * inv;
            acc[j][r] = p;
            __builtin_nontemporal_store(p, arow + wave * 256 + j * 16 + lr);
        }
    }

    // ---- phase 2: ctx = probs @ V ------------------------------------------
    v8f cacc[4] = {};
    #pragma unroll
    for (int jp = 0; jp < 8; ++jp) {
        // stash two 16x16 prob tiles (32 k-columns) as bf16, wave-private
        #pragma unroll
        for (int j2 = 0; j2 < 2; ++j2) {
            int j = jp * 2 + j2;
            #pragma unroll
            for (int r = 0; r < 8; ++r)
                Pst[wave * 640 + (r + 8 * kh) * 40 + j2 * 16 + lr] = (__bf16)acc[j][r];
        }
        // reload as A-fragment (same-wave LDS: hw keeps DS in order)
        v16bf afrag = pack16(*(const v8bf*)&Pst[wave * 640 + lr * 40 + kh * 8],
                             *(const v8bf*)&Pst[wave * 640 + lr * 40 + kh * 8 + 16]);
        int kbase = wave * 256 + jp * 32;
        #pragma unroll
        for (int nt = 0; nt < 4; ++nt) {
            const __bf16* vb =
                Vt + ((long long)(bh * HDd + nt * 16 + lr)) * Sd + kbase + kh * 8;
            v16bf bfrag = pack16(*(const v8bf*)vb, *(const v8bf*)(vb + 16));
            cacc[nt] = __builtin_amdgcn_wmma_f32_16x16x32_bf16(
                false, afrag, false, bfrag, (short)0, cacc[nt], false, false);
        }
    }

    // cross-wave reduce 8 partial ctx blocks, emit merged (b,s,D) bf16
    #pragma unroll
    for (int nt = 0; nt < 4; ++nt)
        #pragma unroll
        for (int r = 0; r < 8; ++r)
            ctxred[wave * 1024 + (r + 8 * kh) * 64 + nt * 16 + lr] = cacc[nt][r];
    __syncthreads();
    #pragma unroll
    for (int i = 0; i < 4; ++i) {
        int idx = t + i * 256;        // 0..1023 = row*64 + col
        int row = idx >> 6, col = idx & 63;
        float s = 0.0f;
        #pragma unroll
        for (int w = 0; w < 8; ++w) s += ctxred[w * 1024 + idx];
        ctxbf[((long long)(bidx * Sd + s0 + row)) * Dd + hidx * HDd + col] = (__bf16)s;
    }
}

// ---------------------------------------------------------------------------
// Elementwise / reshape kernels
// ---------------------------------------------------------------------------
__global__ __launch_bounds__(256) void cvt_bf16(const float* __restrict__ in,
                                                __bf16* __restrict__ out, int n) {
    int i = blockIdx.x * 256 + threadIdx.x;
    if (i < n) out[i] = (__bf16)in[i];
}

__global__ __launch_bounds__(256) void cvt_w_T(const float* __restrict__ in,
                                               __bf16* __restrict__ out,
                                               int K, int N) {
    int i = blockIdx.x * 256 + threadIdx.x;
    if (i < K * N) {
        int k = i / N, n = i - k * N;
        out[(long long)n * K + k] = (__bf16)in[i];
    }
}

__global__ __launch_bounds__(256) void reshape_heads(const float* __restrict__ in,
                                                     __bf16* __restrict__ out) {
    int i = blockIdx.x * 256 + threadIdx.x;   // over BS*D
    int m = i / Dd, n = i - m * Dd;
    int b = m / Sd, s = m - b * Sd;
    int h = n / HDd, d = n - h * HDd;
    out[(((long long)(b * Hd + h) * Sd + s) * HDd) + d] = (__bf16)in[i];
}

__global__ __launch_bounds__(256) void reshape_vT(const float* __restrict__ in,
                                                  __bf16* __restrict__ out) {
    int i = blockIdx.x * 256 + threadIdx.x;   // over BS*D
    int m = i / Dd, n = i - m * Dd;
    int b = m / Sd, s = m - b * Sd;
    int h = n / HDd, d = n - h * HDd;
    out[(((long long)(b * Hd + h) * HDd + d) * Sd) + s] = (__bf16)in[i];
}

__device__ __forceinline__ float blk_red_sum(float v, float* sm) {
    int t = threadIdx.x;
    sm[t] = v; __syncthreads();
    for (int o = 128; o > 0; o >>= 1) { if (t < o) sm[t] += sm[t + o]; __syncthreads(); }
    float r = sm[0]; __syncthreads();
    return r;
}

// x1 = LayerNorm(X + Y)*g + b ; optional fp32 and bf16 outputs. One block/row.
__global__ __launch_bounds__(256) void add_ln(const float* __restrict__ X,
                                              const float* __restrict__ Y,
                                              const float* __restrict__ g,
                                              const float* __restrict__ beta,
                                              float* __restrict__ outf,
                                              __bf16* __restrict__ outbf) {
    __shared__ float sm[256];
    long long base = (long long)blockIdx.x * Dd;
    int t = threadIdx.x;
    float v[4];
    float sum = 0.0f;
    #pragma unroll
    for (int i = 0; i < 4; ++i) {
        int idx = t + i * 256;
        v[i] = X[base + idx] + Y[base + idx];
        sum += v[i];
    }
    sum = blk_red_sum(sum, sm);
    float mu = sum * (1.0f / Dd);
    float var = 0.0f;
    #pragma unroll
    for (int i = 0; i < 4; ++i) { float d = v[i] - mu; var += d * d; }
    var = blk_red_sum(var, sm) * (1.0f / Dd);
    float rstd = rsqrtf(var + 1e-5f);
    #pragma unroll
    for (int i = 0; i < 4; ++i) {
        int idx = t + i * 256;
        float y = (v[i] - mu) * rstd * g[idx] + beta[idx];
        if (outf)  outf[base + idx]  = y;
        if (outbf) outbf[base + idx] = (__bf16)y;
    }
}

// ---------------------------------------------------------------------------
// Host orchestration
// ---------------------------------------------------------------------------
extern "C" void kernel_launch(void* const* d_in, const int* in_sizes, int n_in,
                              void* d_out, int out_size, void* d_ws, size_t ws_size,
                              hipStream_t stream) {
    (void)in_sizes; (void)n_in; (void)out_size; (void)ws_size;

    const float* x    = (const float*)d_in[0];
    const unsigned char* mask = (const unsigned char*)d_in[1];
    const float* Wq = (const float*)d_in[2];  const float* bq = (const float*)d_in[3];
    const float* Wk = (const float*)d_in[4];  const float* bk = (const float*)d_in[5];
    const float* Wv = (const float*)d_in[6];  const float* bv = (const float*)d_in[7];
    const float* Wo = (const float*)d_in[8];  const float* bo = (const float*)d_in[9];
    const float* ln1g = (const float*)d_in[10]; const float* ln1b = (const float*)d_in[11];
    const float* W1 = (const float*)d_in[12]; const float* b1 = (const float*)d_in[13];
    const float* W2 = (const float*)d_in[14]; const float* b2 = (const float*)d_in[15];
    const float* ln2g = (const float*)d_in[16]; const float* ln2b = (const float*)d_in[17];

    float* x2_out   = (float*)d_out;                          // B*S*D
    float* attn_out = x2_out + (size_t)BSd * Dd;              // B*H*S*S

    // Workspace bump allocator (~136 MB total)
    char* wp = (char*)d_ws;
    auto alloc = [&](size_t bytes) -> char* {
        char* p = wp; wp += (bytes + 255) & ~(size_t)255; return p;
    };
    __bf16* x_bf  = (__bf16*)alloc((size_t)BSd * Dd * 2);
    __bf16* WqT   = (__bf16*)alloc((size_t)Dd * Dd * 2);
    __bf16* WkT   = (__bf16*)alloc((size_t)Dd * Dd * 2);
    __bf16* WvT   = (__bf16*)alloc((size_t)Dd * Dd * 2);
    __bf16* WoT   = (__bf16*)alloc((size_t)Dd * Dd * 2);
    __bf16* W1T   = (__bf16*)alloc((size_t)Dd * Fd * 2);
    __bf16* W2T   = (__bf16*)alloc((size_t)Fd * Dd * 2);
    __bf16* Qh    = (__bf16*)alloc((size_t)BSd * Dd * 2);   // (b,h,s,hd)
    __bf16* Kh    = (__bf16*)alloc((size_t)BSd * Dd * 2);   // (b,h,s,hd)
    __bf16* Vt    = (__bf16*)alloc((size_t)BSd * Dd * 2);   // (b,h,hd,s)
    __bf16* ctxbf = (__bf16*)alloc((size_t)BSd * Dd * 2);   // (b,s,D)
    float*  x1f   = (float*)alloc((size_t)BSd * Dd * 4);
    __bf16* x1bf  = (__bf16*)alloc((size_t)BSd * Dd * 2);
    __bf16* h1bf  = (__bf16*)alloc((size_t)BSd * Fd * 2);   // relu(x1@W1+b1)
    float*  Cbuf  = (float*)alloc((size_t)BSd * Dd * 4);    // reused fp32 scratch

    const dim3 blk128(128), blk256(256);
    const int n_xd = BSd * Dd;                  // 4.19M
    const dim3 gEw((n_xd + 255) / 256);

    // 1) Precision conversions
    hipLaunchKernelGGL(cvt_bf16, gEw, blk256, 0, stream, x, x_bf, n_xd);
    hipLaunchKernelGGL(cvt_w_T, dim3((Dd*Dd + 255) / 256), blk256, 0, stream, Wq, WqT, Dd, Dd);
    hipLaunchKernelGGL(cvt_w_T, dim3((Dd*Dd + 255) / 256), blk256, 0, stream, Wk, WkT, Dd, Dd);
    hipLaunchKernelGGL(cvt_w_T, dim3((Dd*Dd + 255) / 256), blk256, 0, stream, Wv, WvT, Dd, Dd);
    hipLaunchKernelGGL(cvt_w_T, dim3((Dd*Dd + 255) / 256), blk256, 0, stream, Wo, WoT, Dd, Dd);
    hipLaunchKernelGGL(cvt_w_T, dim3((Dd*Fd + 255) / 256), blk256, 0, stream, W1, W1T, Dd, Fd);
    hipLaunchKernelGGL(cvt_w_T, dim3((Fd*Dd + 255) / 256), blk256, 0, stream, W2, W2T, Fd, Dd);

    // 2) Q, K, V projections (M=4096, N=1024, K=1024) + head reshapes
    dim3 gProj(Dd / BN, BSd / BM, 1);
    hipLaunchKernelGGL(gemm_bt, gProj, blk128, 0, stream,
        x_bf, WqT, bq, Cbuf, (__bf16*)nullptr, BSd, Dd, Dd, 0);
    hipLaunchKernelGGL(reshape_heads, gEw, blk256, 0, stream, Cbuf, Qh);

    hipLaunchKernelGGL(gemm_bt, gProj, blk128, 0, stream,
        x_bf, WkT, bk, Cbuf, (__bf16*)nullptr, BSd, Dd, Dd, 0);
    hipLaunchKernelGGL(reshape_heads, gEw, blk256, 0, stream, Cbuf, Kh);

    hipLaunchKernelGGL(gemm_bt, gProj, blk128, 0, stream,
        x_bf, WvT, bv, Cbuf, (__bf16*)nullptr, BSd, Dd, Dd, 0);
    hipLaunchKernelGGL(reshape_vT, gEw, blk256, 0, stream, Cbuf, Vt);

    // 3) fused scores -> softmax -> attn write -> ctx (one block per 16 rows)
    hipLaunchKernelGGL(attn_fused, dim3(Sd / 16, BHd), blk256, 0, stream,
        Qh, Kh, Vt, mask, attn_out, ctxbf);

    // 4) output projection: mha = ctx @ Wo + bo
    hipLaunchKernelGGL(gemm_bt, gProj, blk128, 0, stream,
        ctxbf, WoT, bo, Cbuf, (__bf16*)nullptr, BSd, Dd, Dd, 0);

    // 5) x1 = LN(x + mha) -> fp32 (for residual) and bf16 (FFN input)
    hipLaunchKernelGGL(add_ln, dim3(BSd), blk256, 0, stream,
        x, Cbuf, ln1g, ln1b, x1f, x1bf);

    // 6) FFN1: h1 = relu(x1 @ W1 + b1), bf16 out only
    dim3 gFF1(Fd / BN, BSd / BM, 1);
    hipLaunchKernelGGL(gemm_bt, gFF1, blk128, 0, stream,
        x1bf, W1T, b1, (float*)nullptr, h1bf, BSd, Fd, Dd, 1);

    // 7) FFN2: ff = h1 @ W2 + b2
    hipLaunchKernelGGL(gemm_bt, gProj, blk128, 0, stream,
        h1bf, W2T, b2, Cbuf, (__bf16*)nullptr, BSd, Dd, Fd, 0);

    // 8) x2 = LN(x1 + ff) -> d_out
    hipLaunchKernelGGL(add_ln, dim3(BSd), blk256, 0, stream,
        x1f, Cbuf, ln2g, ln2b, x2_out, (__bf16*)nullptr);
}
